// NonCausalSelfAttention_62809601737132
// MI455X (gfx1250) — matrix operand order repaired and checked by reading, not compiled, not run on hardware
//
#include <hip/hip_runtime.h>
#include <hip/hip_bf16.h>

// ---- CDNA5 WMMA types ----
typedef __attribute__((ext_vector_type(16))) __bf16          v16bf;
typedef __attribute__((ext_vector_type(16))) unsigned short  v16u;
typedef __attribute__((ext_vector_type(8)))  float           v8f;

#define BATCH  4
#define SEQ    2048
#define CDIM   768
#define NHEADS 12
#define HS     64

__device__ __forceinline__ unsigned short f2bf(float f) {
    unsigned int u = __float_as_uint(f);
    u += 0x7FFFu + ((u >> 16) & 1u);   // round-to-nearest-even
    return (unsigned short)(u >> 16);
}

// Low 32 bits of a generic pointer into the LDS aperture are the LDS offset
// (ISA 10.2).
__device__ __forceinline__ unsigned lds_off(const void* p) {
    return (unsigned)(unsigned long long)p;
}

// CDNA5 async copy: global -> LDS, 16 bytes per lane, tracked by ASYNCcnt.
__device__ __forceinline__ void async_ld_b128(unsigned dst_lds,
                                              const unsigned short* src) {
    asm volatile("global_load_async_to_lds_b128 %0, %1, off"
                 :: "v"(dst_lds), "v"(src)
                 : "memory");
}
__device__ __forceinline__ void wait_async0() {
    asm volatile("s_wait_asynccnt 0x0" ::: "memory");
}

// A-operand fragment (bf16 16x32, M x K). Per ISA 7.12.2:
// lanes 0-15: row M=lane, K = {0..7 in V0-3, 16..23 in V4-7}
// lanes 16-31: row M=lane-16, K = {8..15, 24..31}
__device__ __forceinline__ v16bf frag_a(const unsigned short* base, int lane,
                                        int stride, int kofs) {
    v16u t;
    const unsigned short* rowp =
        base + (lane & 15) * stride + ((lane & 16) ? 8 : 0) + kofs;
#pragma unroll
    for (int e = 0; e < 16; e += 2) {
        int k = (e < 8) ? e : (e + 8);
        unsigned int pr = *(const unsigned int*)(rowp + k);  // dword-aligned pair
        t[e]     = (unsigned short)(pr & 0xFFFFu);
        t[e + 1] = (unsigned short)(pr >> 16);
    }
    return __builtin_bit_cast(v16bf, t);
}

// B-operand fragment (bf16 32x16, K x N), storage is [n][k] (k contiguous).
// lanes 0-15: col N=lane, K = {0..7, 16..23} ; lanes 16-31: col N=lane-16, K = {8..15, 24..31}
__device__ __forceinline__ v16bf frag_b(const unsigned short* base, int lane,
                                        int stride, int kofs) {
    v16u t;
    const unsigned short* rowp =
        base + (lane & 15) * stride + ((lane & 16) ? 8 : 0) + kofs;
#pragma unroll
    for (int e = 0; e < 16; e += 2) {
        int k = (e < 8) ? e : (e + 8);
        unsigned int pr = *(const unsigned int*)(rowp + k);
        t[e]     = (unsigned short)(pr & 0xFFFFu);
        t[e + 1] = (unsigned short)(pr >> 16);
    }
    return __builtin_bit_cast(v16bf, t);
}

__device__ __forceinline__ v8f wmma_bf16(v16bf a, v16bf b, v8f c) {
    return __builtin_amdgcn_wmma_f32_16x16x32_bf16(false, a, false, b,
                                                   (short)0, c, false, false);
}

__device__ __forceinline__ v8f vzero8() {
    v8f z;
#pragma unroll
    for (int i = 0; i < 8; ++i) z[i] = 0.0f;
    return z;
}

// ---------------- fp32 -> bf16 conversion ----------------
__global__ void cvt_bf16_kernel(const float* __restrict__ src,
                                unsigned short* __restrict__ dst, int n) {
    int i = blockIdx.x * blockDim.x + threadIdx.x;
    if (i < n) dst[i] = f2bf(src[i]);
}

// ---------------- shared GEMM mainloop ----------------
// C[128x128] block: 8 waves as 4(m) x 2(n); each wave = 32x64 = 2x4 fragments.
template <int N>
__device__ __forceinline__ void gemm_tile(const unsigned short* __restrict__ Ag,
                                          const unsigned short* __restrict__ Bg,
                                          unsigned short* As, unsigned short* Bs,
                                          int m0, int n0, v8f acc[2][4]) {
    const int K = CDIM;
    const int tid = threadIdx.x;
    const int lane = tid & 31, wid = tid >> 5;
    const int wm = wid >> 1, wn = wid & 1;
    const unsigned as_base = lds_off(As);

    for (int k0 = 0; k0 < K; k0 += 32) {
        // A tile [128][32] (row-major, u16 stride 40): async global->LDS b128
#pragma unroll
        for (int j = 0; j < 2; ++j) {
            int i = tid + j * 256;               // 512 x 16B transfers
            int row = i >> 2, kq = (i & 3) << 3; // kq = 0,8,16,24 (u16 units)
            async_ld_b128(as_base + row * 80 + kq * 2,
                          Ag + (size_t)(m0 + row) * K + k0 + kq);
        }
        // B tile transposed to [n][k] (stride 40): b128 load + 8 b16 stores
#pragma unroll
        for (int j = 0; j < 2; ++j) {
            int i = tid + j * 256;
            int k = i >> 4, n8 = (i & 15) << 3;
            uint4 d = *(const uint4*)(Bg + (size_t)(k0 + k) * N + n0 + n8);
            unsigned short* bp = Bs + n8 * 40 + k;
            bp[0 * 40] = (unsigned short)d.x; bp[1 * 40] = (unsigned short)(d.x >> 16);
            bp[2 * 40] = (unsigned short)d.y; bp[3 * 40] = (unsigned short)(d.y >> 16);
            bp[4 * 40] = (unsigned short)d.z; bp[5 * 40] = (unsigned short)(d.z >> 16);
            bp[6 * 40] = (unsigned short)d.w; bp[7 * 40] = (unsigned short)(d.w >> 16);
        }
        wait_async0();
        __syncthreads();

        v16bf a0 = frag_a(As + (wm * 32) * 40, lane, 40, 0);
        v16bf a1 = frag_a(As + (wm * 32 + 16) * 40, lane, 40, 0);
#pragma unroll
        for (int nt = 0; nt < 4; ++nt) {
            v16bf bf = frag_b(Bs + (wn * 64 + nt * 16) * 40, lane, 40, 0);
            acc[0][nt] = wmma_bf16(a0, bf, acc[0][nt]);
            acc[1][nt] = wmma_bf16(a1, bf, acc[1][nt]);
        }
        __syncthreads();
    }
}

// ---------------- QKV GEMM: x[8192,768] @ W[768,2304] + b -> q/k/v bf16 ----
__global__ __launch_bounds__(256) void qkv_gemm_kernel(
    const unsigned short* __restrict__ xb, const unsigned short* __restrict__ wa,
    const float* __restrict__ bias, unsigned short* __restrict__ qb,
    unsigned short* __restrict__ kb, unsigned short* __restrict__ vb) {
    __shared__ unsigned short As[128 * 40];
    __shared__ unsigned short Bs[128 * 40];
    const int m0 = blockIdx.x * 128, n0 = blockIdx.y * 128;

    v8f acc[2][4];
#pragma unroll
    for (int i = 0; i < 2; ++i)
#pragma unroll
        for (int j = 0; j < 4; ++j) acc[i][j] = vzero8();

    gemm_tile<3 * CDIM>(xb, wa, As, Bs, m0, n0, acc);

    const int lane = threadIdx.x & 31, wid = threadIdx.x >> 5;
    const int wm = wid >> 1, wn = wid & 1;
    const int l15 = lane & 15, hf = (lane >> 4) & 1;
#pragma unroll
    for (int mt = 0; mt < 2; ++mt)
#pragma unroll
        for (int nt = 0; nt < 4; ++nt)
#pragma unroll
            for (int r = 0; r < 8; ++r) {
                int row = m0 + wm * 32 + mt * 16 + r + hf * 8;   // token index
                int col = n0 + wn * 64 + nt * 16 + l15;          // 0..2303
                float v = acc[mt][nt][r] + bias[col];
                int which = col / CDIM, cc = col % CDIM;
                int h = cc >> 6, d = cc & 63;
                int b = row >> 11, t = row & 2047;
                unsigned short* dst = (which == 0) ? qb : (which == 1) ? kb : vb;
                dst[(((size_t)b * NHEADS + h) * SEQ + t) * HS + d] = f2bf(v);
            }
}

// ---------------- Flash attention: per (b,h), non-causal ----------------
// grid: (SEQ/128, BATCH*NHEADS), 256 threads. Wave w owns 16 query rows.
__global__ __launch_bounds__(256) void attn_kernel(
    const unsigned short* __restrict__ qg, const unsigned short* __restrict__ kg,
    const unsigned short* __restrict__ vg, unsigned short* __restrict__ yg) {
    __shared__ unsigned short Kt[64 * 72];       // [token][dim]
    __shared__ unsigned short Vt[64 * 72];       // [dim][token] (transposed)
    __shared__ unsigned short Pw[8 * 16 * 72];   // per-wave P staging [row][token]

    const int tid = threadIdx.x, wid = tid >> 5, lane = tid & 31;
    const int l15 = lane & 15, hf = (lane >> 4) & 1;
    const int bh = blockIdx.y;
    const int b = bh / NHEADS, h = bh % NHEADS;
    const size_t base = (size_t)bh * SEQ * HS;
    const int q0 = blockIdx.x * 128 + wid * 16;
    const unsigned kt_base = lds_off(Kt);

    // Q fragments (16 rows x 64 dims = two k-chunks), gathered from global
    v16bf qf0 = frag_a(qg + base + (size_t)q0 * HS, lane, HS, 0);
    v16bf qf1 = frag_a(qg + base + (size_t)q0 * HS, lane, HS, 32);

    float mrun[8], lrun[8];
    v8f accO[4];
#pragma unroll
    for (int r = 0; r < 8; ++r) { mrun[r] = -1e30f; lrun[r] = 0.0f; }
#pragma unroll
    for (int nt = 0; nt < 4; ++nt) accO[nt] = vzero8();

    unsigned short* pw = Pw + wid * 16 * 72;

    for (int t0 = 0; t0 < SEQ; t0 += 64) {
        // K tile [tok][d] (stride 72): async global->LDS b128
#pragma unroll
        for (int j = 0; j < 2; ++j) {
            int i = tid + j * 256;                 // 512 x 16B transfers
            int tok = i >> 3, dq = (i & 7) << 3;   // dq = 0,8,...,56
            async_ld_b128(kt_base + tok * 144 + dq * 2,
                          kg + base + (size_t)(t0 + tok) * HS + dq);
        }
        // V tile transposed [d][tok]: b128 load + 8 strided b16 stores
#pragma unroll
        for (int j = 0; j < 2; ++j) {
            int i = tid + j * 256;
            int tok = i >> 3, dq = (i & 7) << 3;
            uint4 d = *(const uint4*)(vg + base + (size_t)(t0 + tok) * HS + dq);
            unsigned short* vp = Vt + dq * 72 + tok;
            vp[0 * 72] = (unsigned short)d.x; vp[1 * 72] = (unsigned short)(d.x >> 16);
            vp[2 * 72] = (unsigned short)d.y; vp[3 * 72] = (unsigned short)(d.y >> 16);
            vp[4 * 72] = (unsigned short)d.z; vp[5 * 72] = (unsigned short)(d.z >> 16);
            vp[6 * 72] = (unsigned short)d.w; vp[7 * 72] = (unsigned short)(d.w >> 16);
        }
        wait_async0();
        __syncthreads();

        // S = (Q K^T) * 1/sqrt(HS) : 4 tiles of 16 keys
        v8f s[4];
#pragma unroll
        for (int nt = 0; nt < 4; ++nt) {
            v8f c = vzero8();
            c = wmma_bf16(qf0, frag_b(Kt + (nt * 16) * 72, lane, 72, 0), c);
            c = wmma_bf16(qf1, frag_b(Kt + (nt * 16) * 72, lane, 72, 32), c);
#pragma unroll
            for (int r = 0; r < 8; ++r) s[nt][r] = c[r] * 0.125f;
        }

        // online softmax (row stats via intra-half shuffles; halves are disjoint)
#pragma unroll
        for (int r = 0; r < 8; ++r) {
            float mx = fmaxf(fmaxf(s[0][r], s[1][r]), fmaxf(s[2][r], s[3][r]));
            mx = fmaxf(mx, __shfl_xor(mx, 1, 32));
            mx = fmaxf(mx, __shfl_xor(mx, 2, 32));
            mx = fmaxf(mx, __shfl_xor(mx, 4, 32));
            mx = fmaxf(mx, __shfl_xor(mx, 8, 32));
            float nm = fmaxf(mrun[r], mx);
            float rs = 0.0f;
#pragma unroll
            for (int nt = 0; nt < 4; ++nt) {
                float p = __expf(s[nt][r] - nm);
                s[nt][r] = p;
                rs += p;
            }
            rs += __shfl_xor(rs, 1, 32);
            rs += __shfl_xor(rs, 2, 32);
            rs += __shfl_xor(rs, 4, 32);
            rs += __shfl_xor(rs, 8, 32);
            float al = __expf(mrun[r] - nm);
            lrun[r] = lrun[r] * al + rs;
            mrun[r] = nm;
#pragma unroll
            for (int nt = 0; nt < 4; ++nt) accO[nt][r] *= al;
        }

        // stage P (bf16) into per-wave LDS as [row][token] (same-wave DS in-order)
#pragma unroll
        for (int nt = 0; nt < 4; ++nt)
#pragma unroll
            for (int r = 0; r < 8; ++r)
                pw[(r + hf * 8) * 72 + nt * 16 + l15] = f2bf(s[nt][r]);

        // O += P @ V : 4 dim-tiles x 2 token-chunks
#pragma unroll
        for (int nt = 0; nt < 4; ++nt) {
#pragma unroll
            for (int kc = 0; kc < 2; ++kc) {
                v16bf pa = frag_a(pw, lane, 72, kc * 32);
                v16bf vb = frag_b(Vt + (nt * 16) * 72, lane, 72, kc * 32);
                accO[nt] = wmma_bf16(pa, vb, accO[nt]);
            }
        }
        __syncthreads();
    }

    // epilogue: normalize, write y in token-major [B][T][C] bf16
#pragma unroll
    for (int nt = 0; nt < 4; ++nt)
#pragma unroll
        for (int r = 0; r < 8; ++r) {
            int t = q0 + r + hf * 8;
            int d = nt * 16 + l15;
            float v = accO[nt][r] / lrun[r];
            yg[((size_t)(b * SEQ + t)) * CDIM + h * HS + d] = f2bf(v);
        }
}

// ---------------- projection GEMM: y[8192,768] @ Wp[768,768] + b -> f32 ----
__global__ __launch_bounds__(256) void proj_gemm_kernel(
    const unsigned short* __restrict__ yb, const unsigned short* __restrict__ wp,
    const float* __restrict__ bias, float* __restrict__ out) {
    __shared__ unsigned short As[128 * 40];
    __shared__ unsigned short Bs[128 * 40];
    const int m0 = blockIdx.x * 128, n0 = blockIdx.y * 128;

    v8f acc[2][4];
#pragma unroll
    for (int i = 0; i < 2; ++i)
#pragma unroll
        for (int j = 0; j < 4; ++j) acc[i][j] = vzero8();

    gemm_tile<CDIM>(yb, wp, As, Bs, m0, n0, acc);

    const int lane = threadIdx.x & 31, wid = threadIdx.x >> 5;
    const int wm = wid >> 1, wn = wid & 1;
    const int l15 = lane & 15, hf = (lane >> 4) & 1;
#pragma unroll
    for (int mt = 0; mt < 2; ++mt)
#pragma unroll
        for (int nt = 0; nt < 4; ++nt)
#pragma unroll
            for (int r = 0; r < 8; ++r) {
                int row = m0 + wm * 32 + mt * 16 + r + hf * 8;
                int col = n0 + wn * 64 + nt * 16 + l15;
                out[(size_t)row * CDIM + col] = acc[mt][nt][r] + bias[col];
            }
}

// ---------------- launch ----------------
extern "C" void kernel_launch(void* const* d_in, const int* in_sizes, int n_in,
                              void* d_out, int out_size, void* d_ws, size_t ws_size,
                              hipStream_t stream) {
    const float* x      = (const float*)d_in[0];
    const float* W_attn = (const float*)d_in[1];
    const float* b_attn = (const float*)d_in[2];
    const float* W_proj = (const float*)d_in[3];
    const float* b_proj = (const float*)d_in[4];
    float* out = (float*)d_out;

    // workspace layout (bf16 elements), ~67.6 MB total
    const size_t nX  = (size_t)BATCH * SEQ * CDIM;   // 6,291,456
    const size_t nWa = (size_t)CDIM * 3 * CDIM;      // 1,769,472
    const size_t nWp = (size_t)CDIM * CDIM;          //   589,824
    unsigned short* xb = (unsigned short*)d_ws;
    unsigned short* wa = xb + nX;
    unsigned short* wp = wa + nWa;
    unsigned short* qb = wp + nWp;
    unsigned short* kb = qb + nX;
    unsigned short* vb = kb + nX;
    unsigned short* yb = vb + nX;

    cvt_bf16_kernel<<<(int)((nX  + 255) / 256), 256, 0, stream>>>(x, xb, (int)nX);
    cvt_bf16_kernel<<<(int)((nWa + 255) / 256), 256, 0, stream>>>(W_attn, wa, (int)nWa);
    cvt_bf16_kernel<<<(int)((nWp + 255) / 256), 256, 0, stream>>>(W_proj, wp, (int)nWp);

    qkv_gemm_kernel<<<dim3(BATCH * SEQ / 128, 3 * CDIM / 128), 256, 0, stream>>>(
        xb, wa, b_attn, qb, kb, vb);

    attn_kernel<<<dim3(SEQ / 128, BATCH * NHEADS), 256, 0, stream>>>(qb, kb, vb, yb);

    proj_gemm_kernel<<<dim3(BATCH * SEQ / 128, CDIM / 128), 256, 0, stream>>>(
        yb, wp, b_proj, out);
}
